// SNNPPO_34084860461502
// MI455X (gfx1250) — matrix operand order repaired and checked by reading, not compiled, not run on hardware
//
#include <hip/hip_runtime.h>
#include <math.h>

typedef __attribute__((ext_vector_type(16))) _Float16 v16h;
typedef __attribute__((ext_vector_type(8)))  _Float16 v8h;
typedef __attribute__((ext_vector_type(8)))  float    v8f;

#define HDIM   256   // hidden size
#define SDIM   64    // input features
#define KIN    128   // 2*SDIM encoder channels
#define AOUT   18    // action dim
#define SEQ    40    // time steps
#define MT     16    // batch rows per workgroup (one WMMA M tile)

// ---------------- fragment loaders -------------------------------------------
__device__ __forceinline__ v16h make16(v8h lo, v8h hi) {
  v16h r;
#pragma unroll
  for (int i = 0; i < 8; ++i) { r[i] = lo[i]; r[i + 8] = hi[i]; }
  return r;
}

// A 16x32 f16 (MxK) from LDS: lanes 0-15 -> M=lane, K 0..15;
//                             lanes 16-31 -> M=lane-16, K 16..31
__device__ __forceinline__ v16h loadA(const _Float16* base, int rowStride,
                                      int m, int kb, int lane) {
  const int kh = (lane >= 16) ? 16 : 0;
  const _Float16* p = base + m * rowStride + kb + kh;
  v8h lo = *(const v8h*)(p);        // ds_load_b128
  v8h hi = *(const v8h*)(p + 8);    // ds_load_b128
  return make16(lo, hi);
}

// B 32x16 f16 (KxN) from LDS, stored as W[N][K] row-major:
// lanes 0-15 -> N=lane, K=kb..kb+15; lanes 16-31 -> N=lane-16, K=kb+16..kb+31
__device__ __forceinline__ v16h loadB(const _Float16* base, int rowStride,
                                      int n, int kb, int lane) {
  const int ko = (lane >= 16) ? 16 : 0;
  const _Float16* p = base + n * rowStride + kb + ko;
  v8h lo = *(const v8h*)(p);
  v8h hi = *(const v8h*)(p + 8);
  return make16(lo, hi);
}

// B fragment built from 16 consecutive f32 in GLOBAL memory (f32 -> f16)
__device__ __forceinline__ v16h cvt16(const float* p) {
  const float4* q = (const float4*)p;       // 16B aligned by construction
  float4 a = q[0], b = q[1], c = q[2], d = q[3];
  v16h r;
  r[0]  = (_Float16)a.x; r[1]  = (_Float16)a.y; r[2]  = (_Float16)a.z; r[3]  = (_Float16)a.w;
  r[4]  = (_Float16)b.x; r[5]  = (_Float16)b.y; r[6]  = (_Float16)b.z; r[7]  = (_Float16)b.w;
  r[8]  = (_Float16)c.x; r[9]  = (_Float16)c.y; r[10] = (_Float16)c.z; r[11] = (_Float16)c.w;
  r[12] = (_Float16)d.x; r[13] = (_Float16)d.y; r[14] = (_Float16)d.z; r[15] = (_Float16)d.w;
  return r;
}

#define WMMA(A, B, C) __builtin_amdgcn_wmma_f32_16x16x32_f16( \
    false, (A), false, (B), (short)0, (C), false, false)

// ------------------------------- kernel --------------------------------------
__global__ void __launch_bounds__(256)
snn_ppo_kernel(const float* __restrict__ x,      // (B, 64)
               const float* __restrict__ w_in,   // (256, 128)
               const float* __restrict__ w_rec,  // (256, 256)
               const float* __restrict__ w_pi,   // (18, 256)
               float* __restrict__ out)          // (B, 18) softmax
{
  extern __shared__ char smem_raw[];
  _Float16* w_pi_s = (_Float16*)smem_raw;             // [ 32][256]  16 KB (pad)
  _Float16* zbuf   = w_pi_s + 32 * HDIM;              // [2][16][256] 16 KB
  _Float16* sxt    = zbuf + 2 * MT * HDIM;            // [16][128]    4 KB
  float*    smax   = (float*)(sxt + MT * KIN);        // [16][32]     2 KB
  // total ~38 KB -> w_rec AND w_in live in VGPRs

  const int tid   = threadIdx.x;
  const int lane  = tid & 31;
  const int wv    = tid >> 5;           // wave id 0..7
  const int bbase = blockIdx.x * MT;

  const int mrow = lane & 15;           // A-row / B-col / C-col index
  const int hi8  = (lane >= 16) ? 8 : 0;
  const int ko   = (lane >= 16) ? 16 : 0;

  // ---- stage w_pi into LDS as f16 (rows 18..31 zero-padded) ----
  for (int idx = tid; idx < 32 * HDIM; idx += 256) {
    int n = idx >> 8;
    w_pi_s[idx] = (n < AOUT) ? (_Float16)w_pi[idx] : (_Float16)0.0f;
  }
  for (int idx = tid; idx < MT * HDIM; idx += 256) zbuf[idx] = (_Float16)0.0f; // z0 = 0

  // ---- loop-invariant weight B-fragments in registers (global f32 -> f16) --
  v16h wrec[HDIM / 32][2];                           // 128 VGPRs
#pragma unroll
  for (int kt = 0; kt < HDIM / 32; ++kt) {
#pragma unroll
    for (int nt = 0; nt < 2; ++nt) {
      const int n = wv * 32 + nt * 16 + mrow;
      wrec[kt][nt] = cvt16(w_rec + n * HDIM + kt * 32 + ko);
    }
  }
  v16h win[KIN / 32][2];                             // 64 VGPRs
#pragma unroll
  for (int kt = 0; kt < KIN / 32; ++kt) {
#pragma unroll
    for (int nt = 0; nt < 2; ++nt) {
      const int n = wv * 32 + nt * 16 + mrow;
      win[kt][nt] = cvt16(w_in + n * KIN + kt * 32 + ko);
    }
  }

  // ---- constant-current LIF encoder state: 8 channels per thread ----
  const int em  = tid >> 4;             // batch row in tile 0..15
  const int ec0 = (tid & 15) * 8;       // channel base 0..120
  float cur[8], venc[8];
#pragma unroll
  for (int j = 0; j < 8; ++j) {
    int c   = ec0 + j;
    float xv = x[(bbase + em) * SDIM + (c & 63)];
    float sc = (c < SDIM) ? 50.0f * xv : -50.0f * xv;
    cur[j]  = fmaxf(sc, 0.0f);
    venc[j] = 0.0f;
  }

  // ---- per-wave LIF cell state (C/D fragment layout): 2 N-tiles of H ----
  v8f vmem[2] = {}, icur[2] = {};
  v8f lv = {}, li = {}, vmax;           // readout state (waves 0,1 only)
#pragma unroll
  for (int e = 0; e < 8; ++e) vmax[e] = -3.4e38f;

  __syncthreads();

  for (int t = 0; t < SEQ; ++t) {
    _Float16* z_old = zbuf + (t & 1) * (MT * HDIM);
    _Float16* z_new = zbuf + ((t + 1) & 1) * (MT * HDIM);

    // --- encoder step -> spike tile sxt (this step's xt) ---
#pragma unroll
    for (int j = 0; j < 8; ++j) {
      float v = 0.9f * venc[j] + 0.1f * cur[j];
      float z = (v > 1.0f) ? 1.0f : 0.0f;
      venc[j] = (1.0f - z) * v;
      sxt[em * KIN + ec0 + j] = (_Float16)z;
    }

    // --- LIF pointwise: v_dec, z_new, v_new; seed accumulator with i_dec ---
    v8f cacc[2];
#pragma unroll
    for (int nt = 0; nt < 2; ++nt) {
#pragma unroll
      for (int e = 0; e < 8; ++e) {
        float vd = 0.9f * vmem[nt][e] + 0.1f * icur[nt][e];
        float z  = (vd > 1.0f) ? 1.0f : 0.0f;
        vmem[nt][e] = (1.0f - z) * vd;
        cacc[nt][e] = 0.8f * icur[nt][e];            // i_dec
        z_new[(e + hi8) * HDIM + (wv * 32 + nt * 16 + mrow)] = (_Float16)z;
      }
    }

    __syncthreads();  // sxt + z_new visible; z_old stable

    // --- i_new = i_dec + xt @ w_in^T + z_old @ w_rec^T  (this wave's 32 cols)
    // Unified 12-tile A stream: tiles 0..3 from sxt, 4..11 from z_old.
    // TWO simultaneously-live buffers (a0, a1) force distinct registers so
    // each WMMA pair waits with the other buffer's loads still in flight.
#define LOAD_ASEQ(i) ((i) < 4 ? loadA(sxt, KIN, mrow, (i) * 32, lane) \
                              : loadA(z_old, HDIM, mrow, ((i) - 4) * 32, lane))
#define BSEL(i, nt)  ((i) < 4 ? win[(i)][nt] : wrec[(i) - 4][nt])

    v16h a0 = LOAD_ASEQ(0);
    v16h a1 = LOAD_ASEQ(1);
#pragma unroll
    for (int i = 0; i < 12; i += 2) {
      cacc[0] = WMMA(a0, BSEL(i, 0), cacc[0]);
      cacc[1] = WMMA(a0, BSEL(i, 1), cacc[1]);
      if (i + 2 < 12) a0 = LOAD_ASEQ(i + 2);
      cacc[0] = WMMA(a1, BSEL(i + 1, 0), cacc[0]);
      cacc[1] = WMMA(a1, BSEL(i + 1, 1), cacc[1]);
      if (i + 3 < 12) a1 = LOAD_ASEQ(i + 3);
    }
#undef LOAD_ASEQ
#undef BSEL
    icur[0] = cacc[0];
    icur[1] = cacc[1];

    // --- readout: i_jump = li + z_new @ w_pi^T ; LI integrator; running max
    if (wv < 2) {                                  // uniform per wave: EXEC stays full
      v8f cpi = li;
      v16h pa0 = loadA(z_new, HDIM, mrow, 0, lane);
      v16h pb0 = loadB(w_pi_s, HDIM, wv * 16 + mrow, 0, lane);
      v16h pa1 = loadA(z_new, HDIM, mrow, 32, lane);
      v16h pb1 = loadB(w_pi_s, HDIM, wv * 16 + mrow, 32, lane);
#pragma unroll
      for (int kt = 0; kt < HDIM / 32; kt += 2) {
        cpi = WMMA(pa0, pb0, cpi);
        if (kt + 2 < HDIM / 32) {
          pa0 = loadA(z_new, HDIM, mrow, (kt + 2) * 32, lane);
          pb0 = loadB(w_pi_s, HDIM, wv * 16 + mrow, (kt + 2) * 32, lane);
        }
        cpi = WMMA(pa1, pb1, cpi);
        if (kt + 3 < HDIM / 32) {
          pa1 = loadA(z_new, HDIM, mrow, (kt + 3) * 32, lane);
          pb1 = loadB(w_pi_s, HDIM, wv * 16 + mrow, (kt + 3) * 32, lane);
        }
      }
#pragma unroll
      for (int e = 0; e < 8; ++e) {
        float ij = cpi[e];
        lv[e]   = 0.9f * lv[e] + 0.1f * ij;
        li[e]   = 0.8f * ij;
        vmax[e] = fmaxf(vmax[e], lv[e]);
      }
    }

    __syncthreads();  // all reads done before next step overwrites LDS tiles
  }

  // ---- gather max-over-time logits, softmax over A=18 ----
  if (wv < 2) {
#pragma unroll
    for (int e = 0; e < 8; ++e)
      smax[(e + hi8) * 32 + (wv * 16 + mrow)] = vmax[e];
  }
  __syncthreads();

  if (tid < MT) {
    const int m = tid;
    float mx = -3.4e38f;
    for (int a = 0; a < AOUT; ++a) mx = fmaxf(mx, smax[m * 32 + a]);
    float ex[AOUT], s = 0.0f;
    for (int a = 0; a < AOUT; ++a) { ex[a] = __expf(smax[m * 32 + a] - mx); s += ex[a]; }
    float inv = 1.0f / s;
    for (int a = 0; a < AOUT; ++a) out[(bbase + m) * AOUT + a] = ex[a] * inv;
  }
}

// ------------------------------ host launch ----------------------------------
extern "C" void kernel_launch(void* const* d_in, const int* in_sizes, int n_in,
                              void* d_out, int out_size, void* d_ws, size_t ws_size,
                              hipStream_t stream) {
  (void)n_in; (void)out_size; (void)d_ws; (void)ws_size;
  const float* x     = (const float*)d_in[0];
  const float* w_in  = (const float*)d_in[1];
  const float* w_rec = (const float*)d_in[2];
  const float* w_pi  = (const float*)d_in[3];
  float* out = (float*)d_out;

  const int B = in_sizes[0] / SDIM;            // 4096
  const int blocks = B / MT;                   // 256 workgroups

  const size_t smem =
      (size_t)(32 * HDIM + 2 * MT * HDIM + MT * KIN) * sizeof(_Float16) +
      (size_t)(MT * 32) * sizeof(float);       // ~38 KB, under the 64 KB default

  snn_ppo_kernel<<<dim3(blocks), dim3(256), smem, stream>>>(x, w_in, w_rec, w_pi, out);
}
